// RBNN_63110249447485
// MI455X (gfx1250) — compile-verified, matrix operand order checked by reading, not run on hardware
//
#include <hip/hip_runtime.h>

// ---------------------------------------------------------------------------
// MI455X (gfx1250, wave32) persistent recurrent-SNN kernel.
//
// Roofline: 1.24 TFLOP total, ~270 MB HBM minimum -> compute/serial-chain
// bound, not bandwidth bound. Weights (19 MB f16) live in L2. 16 persistent
// workgroups (1/WGP) compute all 64 batch rows each step so each weight tile
// is reused across 4 M-tiles; activations are exchanged through L2 in a
// pre-swizzled WMMA-fragment layout so the inner loop is pure
// v_wmma_f32_16x16x32_f16 fed by contiguous b128 loads.
//
// NOTE: k-loops carry unroll(disable) — full unrolling makes clang spill
// VGPRs to scratch inside the WMMA loop (seen in round-2 disasm).
// ---------------------------------------------------------------------------

typedef __attribute__((ext_vector_type(16))) _Float16 v16h;
typedef __attribute__((ext_vector_type(8)))  float    v8f;

#define T_STEPS 1024
#define NWG     16      // persistent workgroups (one per WGP)
#define FF_LEAK 0.5f

// Fragment layout (derived from ISA 16-bit A-matrix table, 16x32 tile):
// tile = 32 lanes x 16 halves, stored contiguously (512 halves = 1 KB).
// lane l (g = l>>4, row = l&15), half j: element [row][kidx(j,g)] where
//   kidx = ((j>>3)&1)*16 + g*8 + ((j>>1)&3)*2 + (j&1)
__device__ __host__ __forceinline__ int kidx_of(int j, int g) {
    return (((j >> 3) & 1) << 4) | (g << 3) | (((j >> 1) & 3) << 1) | (j & 1);
}

__device__ __forceinline__ v8f wmma16x16x32(v16h a, v16h b, v8f c) {
    return __builtin_amdgcn_wmma_f32_16x16x32_f16(
        /*neg_a=*/false, a, /*neg_b=*/false, b,
        /*c_mod=*/(short)0, c, /*reuse_a=*/false, /*reuse_b=*/false);
}

// --------------------------- prep kernels ----------------------------------

// Pack W (row-major [N][K] fp32; B[k][n] = W[n][k]) into f16 fragment tiles:
// dst[nt*nKt + kt][lane*16 + j]. One block per tile.
__global__ void pack_w_kernel(const float* __restrict__ W,
                              _Float16* __restrict__ dst, int K) {
    int nKt = K >> 5;
    int nt = blockIdx.x / nKt, kt = blockIdx.x % nKt;
    _Float16* d = dst + (size_t)blockIdx.x * 512;
    for (int idx = threadIdx.x; idx < 512; idx += 256) {
        int l = idx >> 4, j = idx & 15;
        int k = kt * 32 + kidx_of(j, l >> 4);
        int row = nt * 16 + (l & 15);
        d[idx] = (_Float16)W[(size_t)row * K + k];
    }
}

// Pack inp [T][64][512] fp32 -> frag tiles [t][mt(4)][kt(16)][512] f16.
__global__ void pack_inp_kernel(const float* __restrict__ inp,
                                _Float16* __restrict__ dst) {
    int kt = blockIdx.x & 15;
    int mt = (blockIdx.x >> 4) & 3;
    int t  = blockIdx.x >> 6;
    const float* src = inp + (size_t)t * 64 * 512;
    _Float16* d = dst + (size_t)blockIdx.x * 512;
    for (int idx = threadIdx.x; idx < 512; idx += 256) {
        int l = idx >> 4, j = idx & 15;
        int k = kt * 32 + kidx_of(j, l >> 4);
        int row = mt * 16 + (l & 15);
        d[idx] = (_Float16)src[(size_t)row * 512 + k];
    }
}

// spikes0 = [zeros(64x512) | ones(64x512)] packed as [mt(4)][kt(32)][512];
// also zero the grid-barrier counter.
__global__ void init_state_kernel(_Float16* __restrict__ spk,
                                  unsigned* __restrict__ cnt) {
    if (blockIdx.x == 0 && threadIdx.x == 0) *cnt = 0u;
    int kt = blockIdx.x & 31;
    _Float16* d = spk + (size_t)blockIdx.x * 512;
    for (int idx = threadIdx.x; idx < 512; idx += 256) {
        int j = idx & 15, g = (idx >> 4) >> 4;
        int k = kt * 32 + kidx_of(j, g);
        d[idx] = (k >= 512) ? (_Float16)1.0f : (_Float16)0.0f;
    }
}

// --------------------------- grid barrier ----------------------------------

__device__ __forceinline__ void grid_barrier(unsigned* cnt, unsigned epoch) {
    __threadfence();
    __syncthreads();
    if (threadIdx.x == 0) {
        __hip_atomic_fetch_add(cnt, 1u, __ATOMIC_RELEASE,
                               __HIP_MEMORY_SCOPE_AGENT);
        const unsigned target = epoch * (unsigned)NWG;
        while (__hip_atomic_load(cnt, __ATOMIC_ACQUIRE,
                                 __HIP_MEMORY_SCOPE_AGENT) < target) {
            __builtin_amdgcn_s_sleep(2);
        }
    }
    __syncthreads();
    // CDNA5 cluster barrier: hardware-fast when dispatched as a 16-WG
    // cluster, architected NOP otherwise (ISA 01_flow_sync: out-of-cluster
    // cluster-barrier ops -> S_NOP). Software barrier above guarantees
    // correctness either way.
    __builtin_amdgcn_s_cluster_barrier();
}

// --------------------------- phase helpers ---------------------------------

// Feed-forward phase (pre / post): out = spike(x @ W.T + b), x = [inp_t|spk].
// Per WG: 8 n-tiles (128 cols), per wave: fixed mt, 4 n-tiles (A reused x4).
// k-loop split into inp segment (16 k-tiles) and spike segment (32 k-tiles)
// so the A-pointer select is loop-invariant; unrolling disabled to keep the
// body a compact pipelined (2 A-load + 8 B-load + 4 WMMA) block w/o spills.
__device__ __forceinline__ void phase_ff(
    int t, int wg, int wave, int lane,
    const _Float16* __restrict__ inpP, const _Float16* __restrict__ spkP,
    const _Float16* __restrict__ Wp, const float* __restrict__ bias,
    _Float16* __restrict__ dstP) {
    const int mt    = wave >> 1;
    const int ntg0  = wg * 8 + (wave & 1) * 4;
    const int ncol  = lane & 15;
    const int mrow0 = (lane >> 4) << 3;

    v8f acc[4];
#pragma unroll
    for (int j = 0; j < 4; ++j) {
        float bv = bias[(ntg0 + j) * 16 + ncol];
#pragma unroll
        for (int r = 0; r < 8; ++r) acc[j][r] = bv;
    }

    const _Float16* aInp = inpP + (size_t)t * 32768 + (size_t)mt * 8192 + lane * 16;
    const _Float16* aSpk = spkP + (size_t)mt * 16384 + lane * 16;
    const _Float16* wBase = Wp + (size_t)ntg0 * 48 * 512 + lane * 16;

    // Segment 1: A = inp_t fragments, k-tiles 0..15
#pragma clang loop unroll(disable)
    for (int kt = 0; kt < 16; ++kt) {
        __builtin_prefetch(wBase + (size_t)(kt + 4) * 512, 0, 3);
        v16h a = *(const v16h*)(aInp + (size_t)kt * 512);
#pragma unroll
        for (int j = 0; j < 4; ++j) {
            v16h b = *(const v16h*)(wBase + ((size_t)j * 48 + kt) * 512);
            acc[j] = wmma16x16x32(a, b, acc[j]);
        }
    }
    // Segment 2: A = spike fragments, k-tiles 16..47
#pragma clang loop unroll(disable)
    for (int kt = 0; kt < 32; ++kt) {
        __builtin_prefetch(wBase + (size_t)(20 + kt) * 512, 0, 3);
        v16h a = *(const v16h*)(aSpk + (size_t)kt * 512);
#pragma unroll
        for (int j = 0; j < 4; ++j) {
            v16h b = *(const v16h*)(wBase + ((size_t)j * 48 + 16 + kt) * 512);
            acc[j] = wmma16x16x32(a, b, acc[j]);
        }
    }

    // spike() + scatter into the A-fragment layout of the *next* GEMM
    // (this GEMM's N dimension becomes the next GEMM's K dimension).
#pragma unroll
    for (int j = 0; j < 4; ++j) {
        const int ntg    = ntg0 + j;
        const int klocal = ((ntg & 1) << 4) | ncol;
        const int g      = (klocal >> 3) & 1;
        const int jj     = (((klocal >> 4) & 1) << 3) | (klocal & 7);
        _Float16* base = dstP + (((size_t)mt * 64) + (ntg >> 1)) * 512 + jj;
#pragma unroll
        for (int r = 0; r < 8; ++r) {
            _Float16 s = (acc[j][r] > 0.0f) ? (_Float16)1.0f : (_Float16)0.0f;
            base[(size_t)((g << 4) | (mrow0 + r)) * 16] = s;
        }
    }
}

// Adaptation phase: new_spk = spike(pre @ W_ad.T + b_ad + 0.5*spk), in-place
// spkP update (each WG owns its n-slice). Also emits final_spikes at t=T-1.
__device__ __forceinline__ void phase_ad(
    int t, int wg, int wave, int lane,
    const _Float16* __restrict__ preP, const _Float16* __restrict__ WadP,
    const float* __restrict__ b_ad, _Float16* __restrict__ spkP,
    float* __restrict__ finalSpk) {
    const int mt    = wave >> 1;
    const int ntg0  = wg * 4 + (wave & 1) * 2;
    const int ncol  = lane & 15;
    const int mrow0 = (lane >> 4) << 3;

    v8f acc[2];
#pragma unroll
    for (int j = 0; j < 2; ++j) {
        float bv = b_ad[(ntg0 + j) * 16 + ncol];
#pragma unroll
        for (int r = 0; r < 8; ++r) acc[j][r] = bv;
    }

    const _Float16* ap = preP + (size_t)mt * 32768 + lane * 16;
    const _Float16* wBase = WadP + (size_t)ntg0 * 64 * 512 + lane * 16;
#pragma clang loop unroll(disable)
    for (int kt = 0; kt < 64; ++kt) {
        __builtin_prefetch(wBase + (size_t)(kt + 4) * 512, 0, 3);
        v16h a = *(const v16h*)(ap + (size_t)kt * 512);
#pragma unroll
        for (int j = 0; j < 2; ++j) {
            v16h b = *(const v16h*)(wBase + ((size_t)j * 64 + kt) * 512);
            acc[j] = wmma16x16x32(a, b, acc[j]);
        }
    }

#pragma unroll
    for (int j = 0; j < 2; ++j) {
        const int n      = (ntg0 + j) * 16 + ncol;
        const int klocal = n & 31;
        const int g      = (klocal >> 3) & 1;
        const int jj     = (((klocal >> 4) & 1) << 3) | (klocal & 7);
        _Float16* sb = spkP + (((size_t)mt * 32) + (n >> 5)) * 512 + jj;
#pragma unroll
        for (int r = 0; r < 8; ++r) {
            const int lp = (g << 4) | (mrow0 + r);
            float oldv = (float)sb[(size_t)lp * 16];
            float v    = acc[j][r] + FF_LEAK * oldv;
            _Float16 s = (v > 0.0f) ? (_Float16)1.0f : (_Float16)0.0f;
            sb[(size_t)lp * 16] = s;
            if (t == T_STEPS - 1)
                finalSpk[(size_t)(mt * 16 + mrow0 + r) * 1024 + n] = (float)s;
        }
    }
}

// Output phase: o = post @ W_out.T + b_out -> fp32 d_out[t].
__device__ __forceinline__ void phase_out(
    int t, int wg, int wave, int lane,
    const _Float16* __restrict__ postP, const _Float16* __restrict__ WoutP,
    const float* __restrict__ b_out, float* __restrict__ out) {
    const int mt    = wave >> 1;
    const int ntg   = wg * 2 + (wave & 1);
    const int ncol  = lane & 15;
    const int mrow0 = (lane >> 4) << 3;

    v8f acc;
    {
        float bv = b_out[ntg * 16 + ncol];
#pragma unroll
        for (int r = 0; r < 8; ++r) acc[r] = bv;
    }

    const _Float16* ap = postP + (size_t)mt * 32768 + lane * 16;
    const _Float16* wBase = WoutP + (size_t)ntg * 64 * 512 + lane * 16;
#pragma clang loop unroll(disable)
    for (int kt = 0; kt < 64; ++kt) {
        __builtin_prefetch(wBase + (size_t)(kt + 4) * 512, 0, 3);
        v16h a = *(const v16h*)(ap + (size_t)kt * 512);
        v16h b = *(const v16h*)(wBase + (size_t)kt * 512);
        acc = wmma16x16x32(a, b, acc);
    }

    float* ob = out + (size_t)t * 32768 + (size_t)(mt * 16 + mrow0) * 512 + ntg * 16 + ncol;
#pragma unroll
    for (int r = 0; r < 8; ++r) ob[(size_t)r * 512] = acc[r];
}

// --------------------------- persistent main -------------------------------

__global__ __launch_bounds__(256) void snn_main_kernel(
    const _Float16* __restrict__ inpP,
    const _Float16* __restrict__ WpreP, const _Float16* __restrict__ WadP,
    const _Float16* __restrict__ WpostP, const _Float16* __restrict__ WoutP,
    const float* __restrict__ b_pre, const float* __restrict__ b_ad,
    const float* __restrict__ b_post, const float* __restrict__ b_out,
    _Float16* __restrict__ spkP, _Float16* __restrict__ preP,
    _Float16* __restrict__ postP,
    float* __restrict__ out, float* __restrict__ finalSpk,
    unsigned* __restrict__ cnt) {
    const int lane = threadIdx.x & 31;
    const int wave = threadIdx.x >> 5;
    const int wg   = blockIdx.x;

    unsigned ep = 0;
#pragma clang loop unroll(disable)
    for (int t = 0; t < T_STEPS; ++t) {
        phase_ff(t, wg, wave, lane, inpP, spkP, WpreP, b_pre, preP);
        grid_barrier(cnt, ++ep);
        phase_ad(t, wg, wave, lane, preP, WadP, b_ad, spkP, finalSpk);
        grid_barrier(cnt, ++ep);
        phase_ff(t, wg, wave, lane, inpP, spkP, WpostP, b_post, postP);
        grid_barrier(cnt, ++ep);
        phase_out(t, wg, wave, lane, postP, WoutP, b_out, out);
        grid_barrier(cnt, ++ep);
    }
}

// --------------------------- host entry ------------------------------------

extern "C" void kernel_launch(void* const* d_in, const int* in_sizes, int n_in,
                              void* d_out, int out_size, void* d_ws,
                              size_t ws_size, hipStream_t stream) {
    (void)in_sizes; (void)n_in; (void)out_size; (void)ws_size;

    const float* inp    = (const float*)d_in[0];
    const float* W_pre  = (const float*)d_in[1];
    const float* b_pre  = (const float*)d_in[2];
    const float* W_ad   = (const float*)d_in[3];
    const float* b_ad   = (const float*)d_in[4];
    const float* W_post = (const float*)d_in[5];
    const float* b_post = (const float*)d_in[6];
    const float* W_out  = (const float*)d_in[7];
    const float* b_out  = (const float*)d_in[8];

    char* ws = (char*)d_ws;
    size_t off = 0;
    auto take = [&](size_t bytes) -> char* {
        char* p = ws + off;
        off = (off + bytes + 255) & ~(size_t)255;
        return p;
    };

    unsigned* cnt     = (unsigned*)take(256);
    _Float16* WpreP   = (_Float16*)take((size_t)128 * 48 * 512 * 2);  // 6 MB
    _Float16* WadP    = (_Float16*)take((size_t)64  * 64 * 512 * 2);  // 4 MB
    _Float16* WpostP  = (_Float16*)take((size_t)128 * 48 * 512 * 2);  // 6 MB
    _Float16* WoutP   = (_Float16*)take((size_t)32  * 64 * 512 * 2);  // 2 MB
    _Float16* inpP    = (_Float16*)take((size_t)1024 * 64 * 512 * 2); // 64 MB
    _Float16* spkP    = (_Float16*)take((size_t)4 * 32 * 512 * 2);    // 128 KB
    _Float16* preP    = (_Float16*)take((size_t)4 * 64 * 512 * 2);    // 256 KB
    _Float16* postP   = (_Float16*)take((size_t)4 * 64 * 512 * 2);    // 256 KB

    float* out      = (float*)d_out;
    float* finalSpk = out + (size_t)1024 * 64 * 512;

    init_state_kernel<<<128, 256, 0, stream>>>(spkP, cnt);
    pack_w_kernel<<<128 * 48, 256, 0, stream>>>(W_pre,  WpreP,  1536);
    pack_w_kernel<<<64 * 64,  256, 0, stream>>>(W_ad,   WadP,   2048);
    pack_w_kernel<<<128 * 48, 256, 0, stream>>>(W_post, WpostP, 1536);
    pack_w_kernel<<<32 * 64,  256, 0, stream>>>(W_out,  WoutP,  2048);
    pack_inp_kernel<<<1024 * 4 * 16, 256, 0, stream>>>(inp, inpP);

    snn_main_kernel<<<NWG, 256, 0, stream>>>(
        inpP, WpreP, WadP, WpostP, WoutP,
        b_pre, b_ad, b_post, b_out,
        spkP, preP, postP, out, finalSpk, cnt);
}
